// SpanPruner_76596446757269
// MI455X (gfx1250) — matrix operand order repaired and checked by reading, not compiled, not run on hardware
//
#include <hip/hip_runtime.h>
#include <hip/hip_bf16.h>
#include <stdint.h>

// ---------------- problem constants ----------------
#define T_    2048
#define H_    768
#define W_    20
#define FEAT_ 20
#define U_    1024
#define N_    (T_ * W_)        // 40960 candidate spans
#define SD_   (3 * H_ + FEAT_) // 2324
#define K_TOP 512              // min(MAX_SPANS, int(T*0.4)) = min(512, 819)
#define NS_   65536            // sort padding (pow2 >= N_)
#define KT_   (H_ / 32)        // 24 k-tiles of 32 for WMMA

typedef __attribute__((ext_vector_type(16))) __bf16 v16bf;
typedef __attribute__((ext_vector_type(8)))  __bf16 v8bf;
typedef __attribute__((ext_vector_type(8)))  float  v8f;
typedef __attribute__((ext_vector_type(4)))  unsigned int u32x4;
typedef __attribute__((ext_vector_type(8)))  int i32x8;
typedef __attribute__((ext_vector_type(4)))  int i32x4;

// output layout (floats, reference tuple order, concatenated flat)
constexpr size_t OFF_SPANEMB = 0;
constexpr size_t OFF_IDX     = (size_t)N_ * SD_;
constexpr size_t OFF_STARTS  = OFF_IDX + K_TOP;
constexpr size_t OFF_ENDS    = OFF_STARTS + K_TOP;
constexpr size_t OFF_PEMB    = OFF_ENDS + K_TOP;
constexpr size_t OFF_PSC     = OFF_PEMB + (size_t)K_TOP * SD_;
constexpr size_t OFF_SPK     = OFF_PSC + K_TOP;

__device__ __forceinline__ unsigned short f2bf(float f) {
  unsigned int u = __float_as_uint(f);
  unsigned int r = (u + 0x7FFFu + ((u >> 16) & 1u)) >> 16; // RNE
  return (unsigned short)r;
}

// ---------------- token attention logits: (T,) = H @ w_attn + b ----------------
__global__ __launch_bounds__(256) void k_logits(const float* __restrict__ hs,
                                                const float* __restrict__ wat,
                                                const float* __restrict__ bat,
                                                float* __restrict__ logits) {
  int wave = threadIdx.x >> 5, lane = threadIdx.x & 31;
  int t = blockIdx.x * 8 + wave;
  const float* row = hs + (size_t)t * H_;
  float p = 0.f;
  for (int j = lane; j < H_; j += 32) p += row[j] * wat[j];
  for (int off = 16; off; off >>= 1) p += __shfl_down(p, off, 32);
  if (lane == 0) logits[t] = p + bat[0];
}

// ---------------- fp32 -> bf16 copies ----------------
__global__ void k_cvt_h(const float* __restrict__ hs, unsigned short* __restrict__ hb) {
  int i = blockIdx.x * 256 + threadIdx.x;
  if (i < T_ * H_) hb[i] = f2bf(hs[i]);
}

// Pre-swizzled B for WMMA: Bsw[((z*U + n)*KT + kt)*32 + j] = ws1[(koff[z]+kt*32+j)*U + n]
__global__ void k_cvt_b(const float* __restrict__ ws1, unsigned short* __restrict__ bsw) {
  int i = blockIdx.x * 256 + threadIdx.x;
  const int total = 3 * U_ * H_;
  if (i >= total) return;
  int j  = i & 31;
  int kt = (i >> 5) % KT_;
  int n  = (i / (32 * KT_)) % U_;
  int z  = i / (32 * KT_ * U_);
  int koff = (z == 0) ? 0 : (z == 1 ? H_ : (2 * H_ + FEAT_)); // 0, 768, 1556
  bsw[i] = f2bf(ws1[(size_t)(koff + kt * 32 + j) * U_ + n]);
}

// ---------------- WMMA GEMM: P[z] = Hbf16(2048x768) @ ws1_slice_bf16(768x1024), fp32 acc ---
// block = 256 thr (8 waves, 4x2 grid); block tile 128(M) x 128(N); wave tile 32(M) x 64(N)
__global__ __launch_bounds__(256) void k_gemm(const unsigned short* __restrict__ hb,
                                              const unsigned short* __restrict__ bsw,
                                              float* __restrict__ P) {
  const int z    = blockIdx.z;
  const int wave = threadIdx.x >> 5;
  const int lane = threadIdx.x & 31;
  const int half = lane >> 4;
  const int l15  = lane & 15;
  const int wm   = wave & 3;   // 4 waves along M
  const int wn   = wave >> 2;  // 2 waves along N
  const int m0   = blockIdx.y * 128 + wm * 32;
  const int n0   = blockIdx.x * 128 + wn * 64;
  const unsigned short* bz = bsw + (size_t)z * U_ * H_;
  float* Pz = P + (size_t)z * T_ * U_;

  union FragBF { v16bf v; v8bf h[2]; };
  v8f acc[2][4] = {};

  const size_t arow0 = (size_t)(m0 + l15) * H_;
  const size_t arow1 = (size_t)(m0 + 16 + l15) * H_;
#pragma nounroll
  for (int kt = 0; kt < KT_; ++kt) {
    const int kk = kt * 32;
    FragBF a0, a1;
    a0.h[0] = *reinterpret_cast<const v8bf*>(hb + arow0 + kk + 8 * half);
    a0.h[1] = *reinterpret_cast<const v8bf*>(hb + arow0 + kk + 16 + 8 * half);
    a1.h[0] = *reinterpret_cast<const v8bf*>(hb + arow1 + kk + 8 * half);
    a1.h[1] = *reinterpret_cast<const v8bf*>(hb + arow1 + kk + 16 + 8 * half);
    __builtin_prefetch(hb + arow0 + kk + 32, 0, 1);
    __builtin_prefetch(hb + arow1 + kk + 32, 0, 1);
#pragma unroll
    for (int t = 0; t < 4; ++t) {
      const int n = n0 + t * 16 + l15;
      const unsigned short* bp = bz + (((size_t)n * KT_ + kt) * 32 + 8 * half);
      FragBF b;
      b.h[0] = *reinterpret_cast<const v8bf*>(bp);
      b.h[1] = *reinterpret_cast<const v8bf*>(bp + 16);
      acc[0][t] = __builtin_amdgcn_wmma_f32_16x16x32_bf16(
          false, a0.v, false, b.v, (short)0, acc[0][t], false, false);
      acc[1][t] = __builtin_amdgcn_wmma_f32_16x16x32_bf16(
          false, a1.v, false, b.v, (short)0, acc[1][t], false, false);
    }
  }
#pragma unroll
  for (int u = 0; u < 2; ++u)
#pragma unroll
    for (int t = 0; t < 4; ++t)
#pragma unroll
      for (int r = 0; r < 8; ++r)
        Pz[(size_t)(m0 + u * 16 + r + 8 * half) * U_ + n0 + t * 16 + l15] = acc[u][t][r];
}

// ---------------- per-width precomputes: P3[w] = emb_width[w] @ ws1[1536:1556]; width prior --
__global__ __launch_bounds__(128) void k_width(const float* __restrict__ emb_w,
                                               const float* __restrict__ ws1,
                                               const float* __restrict__ emb_wp,
                                               const float* __restrict__ ww1,
                                               const float* __restrict__ bw1,
                                               const float* __restrict__ ww2,
                                               const float* __restrict__ bw2,
                                               float* __restrict__ P3,
                                               float* __restrict__ wprior) {
  __shared__ float red[128];
  const int w = blockIdx.x, tid = threadIdx.x;
  float part = 0.f;
  for (int n = tid; n < U_; n += 128) {
    float p3 = 0.f, hp = bw1[n];
    for (int f = 0; f < FEAT_; ++f) {
      p3 += emb_w[w * FEAT_ + f] * ws1[(size_t)(2 * H_ + f) * U_ + n];
      hp += emb_wp[w * FEAT_ + f] * ww1[f * U_ + n];
    }
    P3[w * U_ + n] = p3;
    part += fmaxf(hp, 0.f) * ww2[n];
  }
  red[tid] = part;
  __syncthreads();
  for (int off = 64; off; off >>= 1) {
    if (tid < off) red[tid] += red[tid + off];
    __syncthreads();
  }
  if (tid == 0) wprior[w] = red[0] + bw2[0];
}

// ---------------- span embeddings + fused scores (one block per start token) ----------------
__global__ __launch_bounds__(256) void k_span(const float* __restrict__ hs,
                                              const float* __restrict__ logits,
                                              const float* __restrict__ P,
                                              const float* __restrict__ P3,
                                              const float* __restrict__ wprior,
                                              const float* __restrict__ emb_w,
                                              const float* __restrict__ bs1,
                                              const float* __restrict__ ws2,
                                              const float* __restrict__ bs2,
                                              float* __restrict__ span_emb,
                                              float* __restrict__ scores) {
  const int s = blockIdx.x, tid = threadIdx.x;
  __shared__ float red[256];
  const float* P0  = P;
  const float* P1e = P + (size_t)T_ * U_;
  const float* P2a = P + (size_t)2 * T_ * U_;

  float accH[3] = {0.f, 0.f, 0.f};
  float accA[4] = {0.f, 0.f, 0.f, 0.f};
  float Z = 0.f;
  float h0[3], p0[4], bsr[4], wsr[4];
#pragma unroll
  for (int q = 0; q < 3; ++q) h0[q] = hs[(size_t)s * H_ + tid + 256 * q];
#pragma unroll
  for (int q = 0; q < 4; ++q) {
    int n = tid + 256 * q;
    p0[q] = P0[(size_t)s * U_ + n];
    bsr[q] = bs1[n];
    wsr[q] = ws2[n];
  }

  for (int w = 0; w < W_; ++w) {
    const int t = s + w;
    if (t < T_) {
      float ew = __expf(logits[t]);
      Z += ew;
#pragma unroll
      for (int q = 0; q < 3; ++q) accH[q] += ew * hs[(size_t)t * H_ + tid + 256 * q];
#pragma unroll
      for (int q = 0; q < 4; ++q) accA[q] += ew * P2a[(size_t)t * U_ + tid + 256 * q];
    }
    const int e = (t < T_) ? t : (T_ - 1);
    const int widx = e - s;
    const float invZ = 1.f / Z;
    const size_t row = (size_t)(s * W_ + w) * SD_;
#pragma unroll
    for (int q = 0; q < 3; ++q) {
      int i = tid + 256 * q;
      span_emb[row + i]                 = h0[q];
      span_emb[row + H_ + i]            = hs[(size_t)e * H_ + i];
      span_emb[row + 2 * H_ + FEAT_ + i] = accH[q] * invZ;
    }
    if (tid < FEAT_) span_emb[row + 2 * H_ + tid] = emb_w[widx * FEAT_ + tid];

    float part = 0.f;
#pragma unroll
    for (int q = 0; q < 4; ++q) {
      int n = tid + 256 * q;
      float hv = p0[q] + P1e[(size_t)e * U_ + n] + P3[widx * U_ + n] + accA[q] * invZ + bsr[q];
      part += fmaxf(hv, 0.f) * wsr[q];
    }
    __syncthreads();
    red[tid] = part;
    __syncthreads();
    for (int off = 128; off; off >>= 1) {
      if (tid < off) red[tid] += red[tid + off];
      __syncthreads();
    }
    if (tid == 0) scores[s * W_ + w] = red[0] + bs2[0] + wprior[widx];
    __syncthreads();
  }
}

// ---------------- global bitonic sort (descending by score) ----------------
__global__ void k_sort_init(const float* __restrict__ scores, float* __restrict__ key,
                            int* __restrict__ val) {
  int i = blockIdx.x * 256 + threadIdx.x;
  if (i < NS_) {
    key[i] = (i < N_) ? scores[i] : -3.0e38f;
    val[i] = i;
  }
}

__global__ void k_sort_step(float* __restrict__ key, int* __restrict__ val, int kk, int j) {
  int i = blockIdx.x * 256 + threadIdx.x;
  int ixj = i ^ j;
  if (ixj > i) {
    float a = key[i], b = key[ixj];
    bool dirDesc = ((i & kk) == 0);
    if ((a < b) == dirDesc) {
      key[i] = b; key[ixj] = a;
      int va = val[i]; val[i] = val[ixj]; val[ixj] = va;
    }
  }
}

// ---------------- greedy non-crossing NMS (single wave; widths <= 20) ----------------
__global__ __launch_bounds__(32) void k_select(const int* __restrict__ sval,
                                               const int* __restrict__ cs,
                                               const int* __restrict__ ce,
                                               int* __restrict__ selcnt,
                                               int* __restrict__ sellist) {
  __shared__ int s2e[T_];
  __shared__ int e2s[T_];
  __shared__ int sel[K_TOP];
  const int lane = threadIdx.x;
  for (int p = lane; p < T_; p += 32) { s2e[p] = -1; e2s[p] = T_; }
  __syncthreads();
  int cnt = 0;
  for (int i = 0; i < NS_ && cnt < K_TOP; ++i) {
    int idx = sval[i];
    if (idx >= N_) break;
    int s = cs[idx], e = ce[idx];
    int p1 = s + 1 + lane;
    int p2 = s + lane;
    bool c = (p1 <= e && s2e[p1] > e) || (p2 < e && e2s[p2] < s);
    bool crossing = (__ballot(c) != 0ull);
    if (!crossing) {
      if (lane == 0) {
        sel[cnt] = idx;
        if (e > s2e[s]) s2e[s] = e;
        if (s < e2s[e]) e2s[e] = s;
      }
      ++cnt;
    }
    __syncthreads();
  }
  if (lane == 0) selcnt[0] = cnt;
  __syncthreads();
  for (int q = lane; q < cnt; q += 32) sellist[q] = sel[q];
}

// ---------------- sort selected indices ascending, pad with N-1 ----------------
__global__ __launch_bounds__(512) void k_sortsel(const int* __restrict__ selcnt,
                                                 const int* __restrict__ sellist,
                                                 int* __restrict__ pidx,
                                                 float* __restrict__ out) {
  __shared__ int arr[K_TOP];
  const int tid = threadIdx.x;
  const int cnt = selcnt[0];
  arr[tid] = (tid < cnt) ? sellist[tid] : 0x7FFFFFFF;
  __syncthreads();
  for (int kk = 2; kk <= K_TOP; kk <<= 1)
    for (int j = kk >> 1; j > 0; j >>= 1) {
      int ixj = tid ^ j;
      if (ixj > tid) {
        int a = arr[tid], b = arr[ixj];
        bool up = ((tid & kk) == 0); // ascending segments
        if ((a > b) == up) { arr[tid] = b; arr[ixj] = a; }
      }
      __syncthreads();
    }
  int v = arr[tid];
  if (v > N_ - 1) v = N_ - 1;
  pidx[tid] = v;
  out[OFF_IDX + tid] = (float)v;
}

// ---------------- gather pruned outputs via TDM + async LDS->global stores ----------------
// One block per selected span. One wave issues a tensor_load_to_lds moving the whole
// 1 x SD_ (9296 B) span_emb row into LDS (D# per ISA 8.3/8.4: count=1, type=2,
// data_size=4B, tile 2324x1), waits on TENSORcnt, then all threads drain LDS to the
// pruned_emb region with global_store_async_from_lds_b128.
__global__ __launch_bounds__(256) void k_gather(const int* __restrict__ pidx,
                                                const float* __restrict__ scores,
                                                const int* __restrict__ cs,
                                                const int* __restrict__ ce,
                                                const int* __restrict__ spk,
                                                float* __restrict__ out) {
  __shared__ float tile[SD_];                      // 9296 bytes (581 x 16B)
  const int b = blockIdx.x, tid = threadIdx.x;
  const int r = pidx[b];
  const float* src = out + OFF_SPANEMB + (size_t)r * SD_;
  float* dst = out + OFF_PEMB + (size_t)b * SD_;

  if (tid < 32) {
    const unsigned long long ga = (unsigned long long)(const void*)src;
    const unsigned ldsa = (unsigned)(unsigned long long)(void*)&tile[0];
    u32x4 g0;
    g0[0] = 1u;                                    // count=1, user mode
    g0[1] = ldsa;                                  // lds_addr
    g0[2] = (unsigned)(ga & 0xFFFFFFFFu);          // global_addr[31:0]
    g0[3] = (unsigned)((ga >> 32) & 0x01FFFFFFu) | (2u << 30); // addr[56:32] | type=2
    i32x8 g1;
    g1[0] = 0x00020000;                            // data_size=2 (4B), no multicast
    g1[1] = (int)(SD_ << 16);                      // tensor_dim0 = 2324 (lo16 in [31:16])
    g1[2] = (int)(1u << 16);                       // tensor_dim0 hi=0 | tensor_dim1 = 1
    g1[3] = (int)(SD_ << 16);                      // tensor_dim1 hi=0 | tile_dim0 = 2324
    g1[4] = 1;                                     // tile_dim1 = 1, tile_dim2 = 0
    g1[5] = SD_;                                   // tensor_dim0_stride lo32
    g1[6] = 0;                                     // stride hi | tensor_dim1_stride lo
    g1[7] = 0;
    i32x4 gz4 = {0, 0, 0, 0};
    i32x8 gz8 = {0, 0, 0, 0, 0, 0, 0, 0};
    __builtin_amdgcn_tensor_load_to_lds(g0, g1, gz4, gz4, gz8, 0);
    __builtin_amdgcn_s_wait_tensorcnt(0);
  }
  __syncthreads();

  // drain LDS -> global with async stores (581 16-byte chunks)
  const unsigned ldsbase = (unsigned)(unsigned long long)(void*)&tile[0];
  const unsigned long long gdst = (unsigned long long)(void*)dst;
  for (int c = tid; c < SD_ / 4; c += 256) {
    unsigned long long ga = gdst + (unsigned long long)c * 16u;
    unsigned la = ldsbase + (unsigned)c * 16u;
    asm volatile("global_store_async_from_lds_b128 %0, %1, off"
                 :: "v"(ga), "v"(la)
                 : "memory");
  }
  asm volatile("s_wait_asynccnt 0x0" ::: "memory");

  if (tid == 0) {
    int s = cs[r], e = ce[r];
    out[OFF_STARTS + b] = (float)s;
    out[OFF_ENDS + b]   = (float)e;
    out[OFF_PSC + b]    = scores[r];
    out[OFF_SPK + b]    = (float)spk[s];
  }
}

// ---------------- host orchestration ----------------
extern "C" void kernel_launch(void* const* d_in, const int* in_sizes, int n_in,
                              void* d_out, int out_size, void* d_ws, size_t ws_size,
                              hipStream_t stream) {
  const float* hs     = (const float*)d_in[0];
  const int*   cs     = (const int*)d_in[1];
  const int*   ce     = (const int*)d_in[2];
  const int*   spk    = (const int*)d_in[3];
  const float* wat    = (const float*)d_in[4];
  const float* bat    = (const float*)d_in[5];
  const float* ws1    = (const float*)d_in[6];
  const float* bs1    = (const float*)d_in[7];
  const float* ws2    = (const float*)d_in[8];
  const float* bs2    = (const float*)d_in[9];
  const float* ww1    = (const float*)d_in[10];
  const float* bw1    = (const float*)d_in[11];
  const float* ww2    = (const float*)d_in[12];
  const float* bw2    = (const float*)d_in[13];
  const float* emb_w  = (const float*)d_in[14];
  const float* emb_wp = (const float*)d_in[15];
  float* out = (float*)d_out;

  auto au = [](size_t x) { return (x + 255) & ~(size_t)255; };
  char* w = (char*)d_ws;
  float* logits        = (float*)w;          w += au((size_t)T_ * 4);
  unsigned short* hbf  = (unsigned short*)w; w += au((size_t)T_ * H_ * 2);
  unsigned short* bsw  = (unsigned short*)w; w += au((size_t)3 * U_ * H_ * 2);
  float* P             = (float*)w;          w += au((size_t)3 * T_ * U_ * 4);
  float* P3            = (float*)w;          w += au((size_t)W_ * U_ * 4);
  float* wprior        = (float*)w;          w += au((size_t)W_ * 4);
  float* scores        = (float*)w;          w += au((size_t)N_ * 4);
  float* skey          = (float*)w;          w += au((size_t)NS_ * 4);
  int*   sval          = (int*)w;            w += au((size_t)NS_ * 4);
  int*   selcnt        = (int*)w;            w += au(4);
  int*   sellist       = (int*)w;            w += au((size_t)K_TOP * 4);
  int*   pidx          = (int*)w;            w += au((size_t)K_TOP * 4);
  if ((size_t)(w - (char*)d_ws) > ws_size) return;

  // 1) token logits
  k_logits<<<T_ / 8, 256, 0, stream>>>(hs, wat, bat, logits);
  // 2) bf16 conversions (+ B swizzle for WMMA fragment loads)
  k_cvt_h<<<(T_ * H_ + 255) / 256, 256, 0, stream>>>(hs, hbf);
  k_cvt_b<<<(3 * U_ * H_ + 255) / 256, 256, 0, stream>>>(ws1, bsw);
  // 3) P[z] = H @ ws1_slice  (bf16 WMMA, fp32 acc); M=2048,N=1024,K=768, z in {start,end,attn}
  {
    dim3 g(U_ / 128, T_ / 128, 3);
    k_gemm<<<g, 256, 0, stream>>>(hbf, bsw, P);
  }
  // 4) per-width precomputes
  k_width<<<W_, 128, 0, stream>>>(emb_w, ws1, emb_wp, ww1, bw1, ww2, bw2, P3, wprior);
  // 5) span embeddings + fused scores (streams out the 380 MB span_emb once)
  k_span<<<T_, 256, 0, stream>>>(hs, logits, P, P3, wprior, emb_w, bs1, ws2, bs2,
                                 out + OFF_SPANEMB, scores);
  // 6) descending bitonic sort of scores
  k_sort_init<<<NS_ / 256, 256, 0, stream>>>(scores, skey, sval);
  for (int kk = 2; kk <= NS_; kk <<= 1)
    for (int j = kk >> 1; j > 0; j >>= 1)
      k_sort_step<<<NS_ / 256, 256, 0, stream>>>(skey, sval, kk, j);
  // 7) greedy non-crossing selection, 8) order selected indices, 9) TDM gather
  k_select<<<1, 32, 0, stream>>>(sval, cs, ce, selcnt, sellist);
  k_sortsel<<<1, K_TOP, 0, stream>>>(selcnt, sellist, pidx, out);
  k_gather<<<K_TOP, 256, 0, stream>>>(pidx, scores, cs, ce, spk, out);
}
